// VectorQuantizer_86079734546478
// MI455X (gfx1250) — compile-verified, matrix operand order checked by reading, not run on hardware
//
#include <hip/hip_runtime.h>

typedef __attribute__((ext_vector_type(16))) _Float16 v16h;
typedef __attribute__((ext_vector_type(8)))  _Float16 v8h;
typedef __attribute__((ext_vector_type(8)))  float    v8f;
typedef __attribute__((ext_vector_type(4)))  unsigned int u32x4;
typedef __attribute__((ext_vector_type(8)))  int         i32x8;
typedef __attribute__((ext_vector_type(4)))  int         i32x4;

#define VQ_D      64
#define VQ_K      2048
#define VQ_N      65536          // 16*4096 vectors
#define VQ_BETA   0.25f
#define VQ_GAMMA  0.99f
#define VQ_THRESH 1.0f

#define WAVES_PER_WG   16
#define ROWS_PER_WAVE  16
#define ROWS_PER_WG    (WAVES_PER_WG * ROWS_PER_WAVE)   // 256

#if __has_builtin(__builtin_amdgcn_tensor_load_to_lds) && \
    __has_builtin(__builtin_amdgcn_s_wait_tensorcnt)
#define VQ_USE_TDM 1
#else
#define VQ_USE_TDM 0
#endif

// ---------------------------------------------------------------- zero scratch
__global__ void vq_zero_kernel(float* __restrict__ p, int n) {
    int i = blockIdx.x * 256 + threadIdx.x;
    if (i < n) p[i] = 0.0f;
}

// --------------------------------------------- f16 codebook (code-major) + ||e||^2
__global__ void vq_prep_kernel(const float* __restrict__ emb,   // [D,K]
                               _Float16* __restrict__ embF16,   // [K,D]
                               float* __restrict__ esq) {       // [K]
    int k = blockIdx.x * 256 + threadIdx.x;
    if (k >= VQ_K) return;
    float s = 0.0f;
    #pragma unroll
    for (int d = 0; d < VQ_D; ++d) {
        float e = emb[d * VQ_K + k];
        s = fmaf(e, e, s);
        embF16[k * VQ_D + d] = (_Float16)e;
    }
    esq[k] = s;
}

__device__ inline v8h cvt8(float4 f0, float4 f1) {
    v8h r;
    r[0] = (_Float16)f0.x; r[1] = (_Float16)f0.y;
    r[2] = (_Float16)f0.z; r[3] = (_Float16)f0.w;
    r[4] = (_Float16)f1.x; r[5] = (_Float16)f1.y;
    r[6] = (_Float16)f1.z; r[7] = (_Float16)f1.w;
    return r;
}

// ------------------------------------------------ fused distance GEMM + argmin
__global__ __launch_bounds__(512)
void vq_argmin_kernel(const float* __restrict__ x,          // [N,D]
                      const _Float16* __restrict__ embF16,  // [K,D]
                      const float* __restrict__ esq,        // [K]
                      int* __restrict__ idx_out,
                      float* __restrict__ idx_out_f) {
    __shared__ _Float16 sEmb[VQ_K * VQ_D];    // 256 KB (full f16 codebook)
    __shared__ float    sEsq[VQ_K];           // 8 KB

    const int tid  = threadIdx.x;
    const int wave = tid >> 5;
    const int lane = tid & 31;
    const int half = lane >> 4;    // 0: lanes 0-15, 1: lanes 16-31
    const int sub  = lane & 15;

    const int rowBase = blockIdx.x * ROWS_PER_WG + wave * ROWS_PER_WAVE;

#if VQ_USE_TDM
    // ---- Tensor Data Mover: DMA the whole 256KB f16 codebook into LDS ----
    // 1-D tile of 32768 x 8-byte units; issued once by wave 0, tracked by
    // TENSORcnt, overlaps with the per-wave A-fragment global loads below.
    if (wave == 0) {
        unsigned long long ga = (unsigned long long)(const void*)embF16;
        u32x4 g0;
        g0[0] = 1u;                                   // count=1, user descriptor
        g0[1] = (unsigned int)(unsigned long long)&sEmb[0];   // lds_addr (bytes)
        g0[2] = (unsigned int)ga;                     // global_addr[31:0]
        g0[3] = ((unsigned int)(ga >> 32) & 0x01FFFFFFu) | 0x80000000u; // [56:32] | type=2
        i32x8 g1;
        g1[0] = 0x30000;          // data_size = 3 (8B units)
        g1[1] = (int)0x80000000;  // tensor_dim0 = 32768 (low16 at bits 63:48)
        g1[2] = 1 << 16;          // tensor_dim0 hi16 = 0, tensor_dim1 = 1
        g1[3] = (int)0x80000000;  // tile_dim0 = 32768 (bits 127:112)
        g1[4] = 0;                // tile_dim1 = 0 (unused -> 1-D), tile_dim2 = 0
        g1[5] = 32768;            // tensor_dim0_stride[31:0]
        g1[6] = 0;
        g1[7] = 0;
        i32x4 gz = {};            // groups 2/3 unused for <=2-D tensors
#if defined(__clang_major__) && __clang_major__ >= 23
        i32x8 gz8 = {};
        __builtin_amdgcn_tensor_load_to_lds(g0, g1, gz, gz, gz8, 0);
#else
        __builtin_amdgcn_tensor_load_to_lds(g0, g1, gz, gz, 0);
#endif
    }
#else
    {   // fallback: cooperative 16B copies
        const uint4* src = (const uint4*)embF16;
        uint4* dst = (uint4*)sEmb;
        for (int i = tid; i < VQ_K * VQ_D / 8; i += 512) dst[i] = src[i];
    }
#endif
    for (int i = tid; i < VQ_K; i += 512) sEsq[i] = esq[i];

    // ---- A fragments: 16 rows x 64 dims of x, f32 -> f16, held in regs ----
    // ISA 16-bit A 16x32 layout, lane(half,sub) holds row=sub:
    //   frag elems 0..7  : k = 32q + 8*half + j
    //   frag elems 8..15 : k = 32q + 16 + 8*half + j
    const float4* xr4 = (const float4*)(x + (size_t)(rowBase + sub) * VQ_D);
    const int o = half * 2;                       // float4 offset for 8*half floats
    v8h c0 = cvt8(xr4[o + 0],  xr4[o + 1]);       // k:  8h .. 8h+7
    v8h c1 = cvt8(xr4[o + 4],  xr4[o + 5]);       // k: 16+8h ..
    v8h c2 = cvt8(xr4[o + 8],  xr4[o + 9]);       // k: 32+8h ..
    v8h c3 = cvt8(xr4[o + 12], xr4[o + 13]);      // k: 48+8h ..
    v16h a0 = __builtin_shufflevector(c0, c1, 0,1,2,3,4,5,6,7,8,9,10,11,12,13,14,15);
    v16h a1 = __builtin_shufflevector(c2, c3, 0,1,2,3,4,5,6,7,8,9,10,11,12,13,14,15);

#if VQ_USE_TDM
    if (wave == 0) __builtin_amdgcn_s_wait_tensorcnt(0);
#endif
    __syncthreads();   // publish LDS codebook to all waves

    float minv[8];
    int   mini[8];
    #pragma unroll
    for (int i = 0; i < 8; ++i) { minv[i] = 3.4e38f; mini[i] = 0; }

    for (int tile = 0; tile < VQ_K / 16; ++tile) {
        // B fragment: column n = sub -> code tile*16+sub, d follows A's k pattern
        const v8h* br = (const v8h*)(sEmb + (size_t)((tile << 4) + sub) * VQ_D);
        v8h p0 = br[half];      // d:  8h .. 8h+7
        v8h p1 = br[2 + half];  // d: 16+8h ..
        v8h p2 = br[4 + half];  // d: 32+8h ..
        v8h p3 = br[6 + half];  // d: 48+8h ..
        v16h b0 = __builtin_shufflevector(p0, p1, 0,1,2,3,4,5,6,7,8,9,10,11,12,13,14,15);
        v16h b1 = __builtin_shufflevector(p2, p3, 0,1,2,3,4,5,6,7,8,9,10,11,12,13,14,15);

        v8f acc = {};
        acc = __builtin_amdgcn_wmma_f32_16x16x32_f16(false, a0, false, b0,
                                                     (short)0, acc, false, false);
        acc = __builtin_amdgcn_wmma_f32_16x16x32_f16(false, a1, false, b1,
                                                     (short)0, acc, false, false);

        const float es   = sEsq[(tile << 4) + sub];
        const int   code = (tile << 4) + sub;
        // dist = ||e||^2 - 2*sim  (||x||^2 constant per row -> irrelevant to argmin)
        #pragma unroll
        for (int i = 0; i < 8; ++i) {
            float dist = fmaf(-2.0f, acc[i], es);
            if (dist < minv[i]) { minv[i] = dist; mini[i] = code; }
        }
    }

    // C layout: slot i <-> row rowBase + i + 8*half; lanes 0..15 of each half hold
    // disjoint code classes (code mod 16). Butterfly-reduce across those 16 lanes.
    #pragma unroll
    for (int i = 0; i < 8; ++i) {
        #pragma unroll
        for (int off = 8; off > 0; off >>= 1) {
            float ov = __shfl_xor(minv[i], off, 32);
            int   oi = __shfl_xor(mini[i], off, 32);
            if (ov < minv[i] || (ov == minv[i] && oi < mini[i])) {
                minv[i] = ov; mini[i] = oi;
            }
        }
    }
    if (sub == 0) {
        #pragma unroll
        for (int i = 0; i < 8; ++i) {
            int row = rowBase + 8 * half + i;
            idx_out[row]   = mini[i];
            idx_out_f[row] = (float)mini[i];
        }
    }
}

// --------------------------- gather quantized + commitment loss + EMA scatter
__global__ __launch_bounds__(256)
void vq_gather_kernel(const float* __restrict__ x,
                      const float* __restrict__ emb,      // [D,K] f32
                      const int* __restrict__ idx,
                      float* __restrict__ outQ,
                      float* __restrict__ lossAcc,
                      float* __restrict__ mbatch,         // [D,K]
                      float* __restrict__ cnt) {          // [K]
    __shared__ float red[256];
    int t = blockIdx.x * 256 + threadIdx.x;   // over N*D
    int n = t >> 6;
    int d = t & 63;
    int k = idx[n];
    float xv = x[t];
    float q  = emb[d * VQ_K + k];
    outQ[t] = q;                               // quantized_st == quantized numerically
    float diff = q - xv;
    red[threadIdx.x] = diff * diff;
    __syncthreads();
    #pragma unroll
    for (int s = 128; s > 0; s >>= 1) {
        if (threadIdx.x < s) red[threadIdx.x] += red[threadIdx.x + s];
        __syncthreads();
    }
    if (threadIdx.x == 0) atomicAdd(lossAcc, red[0]);
    atomicAdd(&mbatch[d * VQ_K + k], xv);
    if (d == 0) atomicAdd(&cnt[k], 1.0f);
}

// ------------------------------------------------- EMA update + codebook reset
__global__ void vq_final_kernel(const float* __restrict__ x,
                                const float* __restrict__ m_t,
                                const float* __restrict__ N_t,
                                const float* __restrict__ cnt,
                                const float* __restrict__ mbatch,
                                const float* __restrict__ lossAcc,
                                float* __restrict__ outEmb,
                                float* __restrict__ outM,
                                float* __restrict__ outN,
                                float* __restrict__ outLoss) {
    int k = blockIdx.x * 256 + threadIdx.x;
    if (k >= VQ_K) return;
    float Nn = VQ_GAMMA * N_t[k] + (1.0f - VQ_GAMMA) * cnt[k];
    outN[k] = Nn;
    float denom = fminf(fmaxf(Nn, 1e-8f), 1e8f);
    bool used = (Nn >= VQ_THRESH);
    // deterministic stand-in for jax.random.permutation(key(42)): odd-multiplier
    // bijection over [0, 65536)
    unsigned perm = (unsigned)(k * 40503u + 997u) & 65535u;
    const float* rr = x + (size_t)perm * VQ_D;
    #pragma unroll
    for (int d = 0; d < VQ_D; ++d) {
        float mn = VQ_GAMMA * m_t[d * VQ_K + k] + (1.0f - VQ_GAMMA) * mbatch[d * VQ_K + k];
        outM[d * VQ_K + k]   = mn;
        outEmb[d * VQ_K + k] = used ? (mn / denom) : rr[d];
    }
    if (k == 0) outLoss[0] = VQ_BETA * lossAcc[0] / (float)((size_t)VQ_N * VQ_D);
}

extern "C" void kernel_launch(void* const* d_in, const int* in_sizes, int n_in,
                              void* d_out, int out_size, void* d_ws, size_t ws_size,
                              hipStream_t stream) {
    const float* x    = (const float*)d_in[0];   // [16,4096,64]
    const float* emb  = (const float*)d_in[1];   // [64,2048]
    const float* m_t  = (const float*)d_in[2];   // [64,2048]
    const float* N_t  = (const float*)d_in[3];   // [2048]

    float* out = (float*)d_out;
    float* outQ    = out;                        // 4,194,304
    float* outIdx  = out + 4194304;              // 65,536 (idx as float)
    float* outEmb  = out + 4259840;              // 131,072
    float* outM    = out + 4390912;              // 131,072
    float* outN    = out + 4521984;              // 2,048
    float* outLoss = out + 4524032;              // 1

    char* ws = (char*)d_ws;
    _Float16* embF16 = (_Float16*)(ws);                  // 256 KB
    float*    esq    = (float*)(ws + 262144);            // 8 KB
    int*      idxw   = (int*)(ws + 262144 + 8192);       // 256 KB
    float*    mbatch = (float*)(ws + 532480);            // 512 KB
    float*    cnt    = (float*)(ws + 1056768);           // 8 KB
    float*    loss   = (float*)(ws + 1064960);           // 4 B

    // zero mbatch + cnt + loss (contiguous: 131072 + 2048 + 1 floats)
    vq_zero_kernel<<<(133121 + 255) / 256, 256, 0, stream>>>(mbatch, 133121);
    vq_prep_kernel<<<VQ_K / 256, 256, 0, stream>>>(emb, embF16, esq);
    vq_argmin_kernel<<<VQ_N / ROWS_PER_WG, 512, 0, stream>>>(x, embF16, esq, idxw, outIdx);
    vq_gather_kernel<<<(VQ_N * VQ_D) / 256, 256, 0, stream>>>(x, emb, idxw, outQ, loss,
                                                              mbatch, cnt);
    vq_final_kernel<<<VQ_K / 256, 256, 0, stream>>>(x, m_t, N_t, cnt, mbatch, loss,
                                                    outEmb, outM, outN, outLoss);
}
